// Model_57234734186890
// MI455X (gfx1250) — matrix operand order repaired and checked by reading, not compiled, not run on hardware
//
#include <hip/hip_runtime.h>

// ---------------------------------------------------------------------------
// Problem constants (from reference): T,B,I,H,NH,O = 2048,256,64,128,512,64
// ---------------------------------------------------------------------------
constexpr int T_STEPS = 2048;
constexpr int BSZ     = 256;
constexpr int IDIM    = 64;    // input size
constexpr int HDIM    = 128;   // hidden size
constexpr int NHID    = 512;   // MLP hidden
constexpr int ODIM    = 64;    // output size

typedef __bf16 bf16_t;
typedef __attribute__((ext_vector_type(16))) __bf16 v16bf;
typedef __attribute__((ext_vector_type(8)))  float  v8f;
typedef __attribute__((ext_vector_type(4)))  float  v4f;

union FragAB { v16bf v; unsigned u[8]; };

__device__ __forceinline__ v8f wmma_bf16(v16bf a, v16bf b, v8f c) {
    // D = A(16x32 bf16) * B(32x16 bf16) + C(16x16 f32)
    return __builtin_amdgcn_wmma_f32_16x16x32_bf16(
        /*neg_a=*/false, a, /*neg_b=*/false, b,
        /*c_mod=*/(short)0, c, /*reuse_a=*/false, /*reuse_b=*/false);
}

// A-matrix 16x32 bf16 fragment from a row-major bf16 buffer (row stride in elems).
// Contiguous k order used here: lanes 0-15 -> M=lane, K = 0..15; lanes 16-31 -> K = 16..31.
__device__ __forceinline__ v16bf load_frag_A(const bf16_t* base, int stride,
                                             int k0, int lane) {
    const int m = lane & 15, half = lane >> 4;
    const bf16_t* p = base + m * stride + k0 + half * 16;
    FragAB f;
#pragma unroll
    for (int j = 0; j < 4; ++j) f.u[j]     = *(const unsigned*)(p + 2 * j);
#pragma unroll
    for (int j = 0; j < 4; ++j) f.u[4 + j] = *(const unsigned*)(p + 8 + 2 * j);
    return f.v;
}

// B-matrix 32x16 bf16 fragment with B(k,n) = W[n0+n][k0+k], W row-major [N][K].
// Contiguous k order used here: n = lane%16; lanes 0-15 hold K=0..15, lanes 16-31 K=16..31.
__device__ __forceinline__ v16bf load_frag_B(const bf16_t* W, int ldk,
                                             int n0, int k0, int lane) {
    const int n = lane & 15, half = lane >> 4;
    const bf16_t* p = W + (size_t)(n0 + n) * ldk + k0 + half * 16;
    FragAB f;
#pragma unroll
    for (int j = 0; j < 8; ++j) f.u[j] = *(const unsigned*)(p + 2 * j);
    return f.v;
}

// Fast transcendentals: v_exp_f32 + v_rcp_f32 (TRANS ops that co-execute with
// WMMA), avoiding precise div_scale/fma division chains on the serial path.
__device__ __forceinline__ float fsig(float v) {
    return __builtin_amdgcn_rcpf(1.f + __expf(-v));
}
__device__ __forceinline__ float ftanh(float v) {
    // tanh(x) = 1 - 2/(exp(2x)+1); rcp(inf)=0 handles overflow -> +/-1.
    return 1.f - 2.f * __builtin_amdgcn_rcpf(__expf(2.f * v) + 1.f);
}

// Straight-line stage: convert 4 consecutive fp32 -> 4 bf16, one 8B LDS store.
__device__ __forceinline__ void stage4(bf16_t* __restrict__ dst,
                                       const float* __restrict__ src, int idx4) {
    const v4f v = *(const v4f*)(src + idx4 * 4);
    bf16_t tmp[4];
#pragma unroll
    for (int j = 0; j < 4; ++j) tmp[j] = (bf16_t)v[j];
    *(unsigned long long*)(dst + idx4 * 4) = *(const unsigned long long*)tmp;
}

// ---------------------------------------------------------------------------
// Kernel 0: convert all weights fp32 -> bf16 into workspace (once per launch).
// ws element layout (bf16): [0,32768) W_ih | [32768,98304) W_hh |
//                           [98304,163840) W1 | [163840,196608) W2 | hs...
// ---------------------------------------------------------------------------
__global__ void cvt_weights_kernel(const float* __restrict__ wih,
                                   const float* __restrict__ whh,
                                   const float* __restrict__ w1,
                                   const float* __restrict__ w2,
                                   bf16_t* __restrict__ dst) {
    int i = blockIdx.x * blockDim.x + threadIdx.x;
    if (i < 32768)       dst[i] = (bf16_t)wih[i];
    else if (i < 98304)  dst[i] = (bf16_t)whh[i - 32768];
    else if (i < 163840) dst[i] = (bf16_t)w1[i - 98304];
    else if (i < 196608) dst[i] = (bf16_t)w2[i - 163840];
}

// ---------------------------------------------------------------------------
// Kernel 1: LSTM scan. One workgroup (8 waves) per 16-batch-row tile -> 16 WGs,
// no cross-workgroup sync needed (recurrence is independent across batch).
// Wave w owns gate-column group j = w (cols w*16..w*16+15 of each of i/f/g/o).
// Weights live in register fragments for all 2048 steps. Double-buffered LDS
// (x staged one step ahead) -> exactly ONE workgroup barrier per timestep.
// ---------------------------------------------------------------------------
__global__ __launch_bounds__(256) void lstm_scan_kernel(
    const float*  __restrict__ x,       // [T,B,I] fp32
    const float*  __restrict__ b_ih,    // [4H]
    const float*  __restrict__ b_hh,    // [4H]
    const bf16_t* __restrict__ wih_bf,  // [4H,I]
    const bf16_t* __restrict__ whh_bf,  // [4H,H]
    bf16_t*       __restrict__ hs_bf,   // [T,B,H] out (workspace)
    float*        __restrict__ out)     // d_out: [...| hT B*H | cT B*H]
{
    __shared__ bf16_t x_sh[2][16 * IDIM];   // x_t batch tile (bf16), dbl-buffered
    __shared__ bf16_t h_sh[2][16 * HDIM];   // h_t batch tile (bf16), dbl-buffered

    const int lane = threadIdx.x & 31;
    const int wave = threadIdx.x >> 5;
    const int b0   = blockIdx.x * 16;
    const int tid  = threadIdx.x;

    // Prologue (straight-line): h0 = 0 (2048 elems = 256 thr x 8) and stage x_0
    // (1024 elems = 256 thr x 4).
    *(unsigned long long*)(&h_sh[0][tid * 8])     = 0ull;
    *(unsigned long long*)(&h_sh[0][tid * 8 + 4]) = 0ull;
    stage4(x_sh[0], x + (size_t)b0 * IDIM, tid);

    // Hoist weight fragments into registers (held across all 2048 steps).
    v16bf wih_f[4][2];   // [gate][k-chunk of 32 over I=64]
    v16bf whh_f[4][4];   // [gate][k-chunk of 32 over H=128]
    float bias[4];
#pragma unroll
    for (int g = 0; g < 4; ++g) {
        const int n0 = g * HDIM + wave * 16;  // gate g, column group `wave`
#pragma unroll
        for (int kc = 0; kc < 2; ++kc)
            wih_f[g][kc] = load_frag_B(wih_bf, IDIM, n0, kc * 32, lane);
#pragma unroll
        for (int kc = 0; kc < 4; ++kc)
            whh_f[g][kc] = load_frag_B(whh_bf, HDIM, n0, kc * 32, lane);
        const int gc = n0 + (lane & 15);
        bias[g] = b_ih[gc] + b_hh[gc];
    }

    const v8f vzero = {0.f,0.f,0.f,0.f,0.f,0.f,0.f,0.f};
    v8f c_st = vzero;  // fp32 cell state, 16x16 tile (b x hcol-group)

    const int n    = lane & 15;
    const int half = lane >> 4;
    const int hcol = wave * 16 + n;

    __syncthreads();  // h0 / x0 visible

    for (int t = 0; t < T_STEPS; ++t) {
        const int cur = t & 1, nxt = cur ^ 1;

        // A fragments from current buffers (shared across the 4 gate tiles).
        v16bf ax[2], ah[4];
#pragma unroll
        for (int kc = 0; kc < 2; ++kc)
            ax[kc] = load_frag_A(x_sh[cur], IDIM, kc * 32, lane);
#pragma unroll
        for (int kc = 0; kc < 4; ++kc)
            ah[kc] = load_frag_A(h_sh[cur], HDIM, kc * 32, lane);

        // Stage x_{t+1} into the other buffer (straight-line, no conflict with
        // readers) and prefetch x_{t+2} toward L2 (global_prefetch_b8).
        if (t + 1 < T_STEPS) {
            const float* xn = x + (size_t)(t + 1) * BSZ * IDIM + (size_t)b0 * IDIM;
            stage4(x_sh[nxt], xn, tid);
            if (t + 2 < T_STEPS)
                __builtin_prefetch(xn + BSZ * IDIM + tid * 4, 0, 3);
        }

        // gates[g] = x_t @ W_ih^T + h_t @ W_hh^T   (bias added below)
        v8f acc[4];
#pragma unroll
        for (int g = 0; g < 4; ++g) {
            v8f a = vzero;
#pragma unroll
            for (int kc = 0; kc < 2; ++kc) a = wmma_bf16(ax[kc], wih_f[g][kc], a);
#pragma unroll
            for (int kc = 0; kc < 4; ++kc) a = wmma_bf16(ah[kc], whh_f[g][kc], a);
            acc[g] = a;
        }

        // Elementwise LSTM cell (PyTorch gate order i,f,g,o), fp32.
        float hv[8], cv[8];
#pragma unroll
        for (int r = 0; r < 8; ++r) {
            float iv = fsig (acc[0][r] + bias[0]);
            float fv = fsig (acc[1][r] + bias[1]);
            float gv = ftanh(acc[2][r] + bias[2]);
            float ov = fsig (acc[3][r] + bias[3]);
            float cn = fv * c_st[r] + iv * gv;
            c_st[r] = cn;
            cv[r] = cn;
            hv[r] = ov * ftanh(cn);
        }

        // Write h_t: LDS next-buffer (for step t+1) + global hs (bf16).
        const size_t hs_base = (size_t)t * BSZ * HDIM + (size_t)b0 * HDIM;
#pragma unroll
        for (int r = 0; r < 8; ++r) {
            const int m = r + 8 * half;
            const bf16_t hb = (bf16_t)hv[r];
            h_sh[nxt][m * HDIM + hcol] = hb;
            hs_bf[hs_base + (size_t)m * HDIM + hcol] = hb;
        }

        if (t == T_STEPS - 1) {
            const size_t baseh = (size_t)T_STEPS * BSZ * ODIM;
            const size_t basec = baseh + (size_t)BSZ * HDIM;
#pragma unroll
            for (int r = 0; r < 8; ++r) {
                const int m = r + 8 * half;
                out[baseh + (size_t)(b0 + m) * HDIM + hcol] = hv[r];
                out[basec + (size_t)(b0 + m) * HDIM + hcol] = cv[r];
            }
        }

        __syncthreads();  // single barrier per step: nxt buffers now consistent
    }
}

// ---------------------------------------------------------------------------
// Kernel 2: fused MLP head per 16-row block of [T*B, H]:
//   y1 = relu(hs @ W1^T + b1) (in LDS, bf16), out = y1 @ W2^T + b2 (fp32).
// ---------------------------------------------------------------------------
__global__ __launch_bounds__(256) void mlp_head_kernel(
    const bf16_t* __restrict__ hs_bf,  // [T*B, H]
    const bf16_t* __restrict__ w1_bf,  // [NH, H]
    const bf16_t* __restrict__ w2_bf,  // [O, NH]
    const float*  __restrict__ b1,     // [NH]
    const float*  __restrict__ b2,     // [O]
    float*        __restrict__ out)    // [T*B, O]
{
    __shared__ bf16_t hs_sh[16 * HDIM];  // 4 KB
    __shared__ bf16_t y1_sh[16 * NHID];  // 16 KB

    const int lane = threadIdx.x & 31;
    const int wave = threadIdx.x >> 5;
    const int tid  = threadIdx.x;
    const size_t row0 = (size_t)blockIdx.x * 16;

    // Stage 16 hs rows: 1024 dwords = 256 thr x 4 -> one b128 copy per thread.
    {
        typedef __attribute__((ext_vector_type(4))) unsigned u4;
        const u4* src = (const u4*)(hs_bf + row0 * HDIM);
        ((u4*)hs_sh)[tid] = src[tid];
    }
    __syncthreads();

    v16bf a[4];
#pragma unroll
    for (int kc = 0; kc < 4; ++kc)
        a[kc] = load_frag_A(hs_sh, HDIM, kc * 32, lane);

    const v8f vzero = {0.f,0.f,0.f,0.f,0.f,0.f,0.f,0.f};
    const int n    = lane & 15;
    const int half = lane >> 4;

    // Phase 1: y1 tiles; wave w owns n-tiles 4w..4w+3 of NH=512.
#pragma unroll
    for (int tt = 0; tt < 4; ++tt) {
        const int n0 = (wave * 4 + tt) * 16;
        v8f acc = vzero;
#pragma unroll
        for (int kc = 0; kc < 4; ++kc)
            acc = wmma_bf16(a[kc], load_frag_B(w1_bf, HDIM, n0, kc * 32, lane), acc);
        const float bb = b1[n0 + n];
#pragma unroll
        for (int r = 0; r < 8; ++r) {
            float v = acc[r] + bb;
            v = v > 0.f ? v : 0.f;  // relu
            y1_sh[(r + 8 * half) * NHID + n0 + n] = (bf16_t)v;
        }
    }
    __syncthreads();

    // Phase 2: out tiles (O=64 -> 4 tiles); waves 0..3, K=512 (16 chunks).
    if (wave < 4) {
        const int n0 = wave * 16;
        v8f acc = vzero;
#pragma unroll
        for (int kc = 0; kc < 16; ++kc) {
            v16bf aa = load_frag_A(y1_sh, NHID, kc * 32, lane);
            acc = wmma_bf16(aa, load_frag_B(w2_bf, NHID, n0, kc * 32, lane), acc);
        }
        const float bb = b2[n0 + n];
#pragma unroll
        for (int r = 0; r < 8; ++r)
            out[(row0 + r + 8 * half) * ODIM + n0 + n] = acc[r] + bb;
    }
}

// ---------------------------------------------------------------------------
extern "C" void kernel_launch(void* const* d_in, const int* in_sizes, int n_in,
                              void* d_out, int out_size, void* d_ws, size_t ws_size,
                              hipStream_t stream) {
    (void)in_sizes; (void)n_in; (void)out_size; (void)ws_size;

    const float* x    = (const float*)d_in[0];
    const float* W_ih = (const float*)d_in[1];
    const float* W_hh = (const float*)d_in[2];
    const float* b_ih = (const float*)d_in[3];
    const float* b_hh = (const float*)d_in[4];
    const float* W1   = (const float*)d_in[5];
    const float* b1   = (const float*)d_in[6];
    const float* W2   = (const float*)d_in[7];
    const float* b2   = (const float*)d_in[8];
    float* out = (float*)d_out;

    bf16_t* wsb    = (bf16_t*)d_ws;
    bf16_t* wih_bf = wsb;
    bf16_t* whh_bf = wsb + 32768;
    bf16_t* w1_bf  = wsb + 98304;
    bf16_t* w2_bf  = wsb + 163840;
    bf16_t* hs_bf  = wsb + 196608;   // [T*B*H] bf16 = 134 MB

    cvt_weights_kernel<<<768, 256, 0, stream>>>(W_ih, W_hh, W1, W2, wsb);
    lstm_scan_kernel<<<BSZ / 16, 256, 0, stream>>>(
        x, b_ih, b_hh, wih_bf, whh_bf, hs_bf, out);
    mlp_head_kernel<<<(T_STEPS * BSZ) / 16, 256, 0, stream>>>(
        hs_bf, w1_bf, w2_bf, b1, b2, out);
}
